// KCNetClassify_40759239639788
// MI455X (gfx1250) — matrix operand order repaired and checked by reading, not compiled, not run on hardware
//
#include <hip/hip_runtime.h>
#include <hip/hip_bf16.h>
#include <math.h>

typedef __attribute__((ext_vector_type(16))) _Float16 v16h;
typedef __attribute__((ext_vector_type(8)))  float    v8f;

#define NCOL 16384   // B*N = 8*2048
#define NPB  2048    // points per batch
#define NEG_BIG -3.0e38f

__device__ __forceinline__ float atomicMaxF(float* addr, float val) {
    int* ai = (int*)addr;
    int old = __float_as_int(*addr);
    while (val > __int_as_float(old)) {
        int assumed = old;
        old = atomicCAS(ai, assumed, __float_as_int(val));
        if (old == assumed) break;
    }
    return __int_as_float(old);
}

// ---------------------------------------------------------------- fill -----
__global__ void fillf_kernel(float* p, int n, float v) {
    int i = blockIdx.x * blockDim.x + threadIdx.x;
    if (i < n) p[i] = v;
}

// ----------------------------------------------------------------- KNN -----
// block = 128 threads (4 waves), each wave owns one query row.
// grid = (512, 8)  -> 512*4 = 2048 rows per batch.
__global__ void knn_kernel(const float* __restrict__ pts, int* __restrict__ knn) {
    __shared__ float px[NPB], py[NPB], pz[NPB];
    __shared__ float d2[4][NPB];
    const int b = blockIdx.y;
    const float* P = pts + b * 3 * NPB;
    for (int i = threadIdx.x; i < NPB; i += 128) {
        px[i] = P[i]; py[i] = P[NPB + i]; pz[i] = P[2 * NPB + i];
    }
    __syncthreads();
    const int wave = threadIdx.x >> 5, lane = threadIdx.x & 31;
    const int n = blockIdx.x * 4 + wave;
    const float qx = px[n], qy = py[n], qz = pz[n];
    for (int i = lane; i < NPB; i += 32) {
        float dx = px[i] - qx, dy = py[i] - qy, dz = pz[i] - qz;
        d2[wave][i] = dx * dx + dy * dy + dz * dz;
    }
    __syncthreads();
    const int outbase = (b * NPB + n) * 17;
    for (int t = 0; t < 17; ++t) {
        float bv = 3.4e38f; int bi = 0x7fffffff;
        for (int i = lane; i < NPB; i += 32) {
            float v = d2[wave][i];
            if (v < bv || (v == bv && i < bi)) { bv = v; bi = i; }
        }
        #pragma unroll
        for (int off = 16; off; off >>= 1) {
            float ov = __shfl_xor(bv, off, 32);
            int   oi = __shfl_xor(bi, off, 32);
            if (ov < bv || (ov == bv && oi < bi)) { bv = ov; bi = oi; }
        }
        if (lane == 0) { knn[outbase + t] = bi; d2[wave][bi] = 3.4e38f; }
        __syncthreads();
    }
}

// ------------------------------------------------------ kernel correlation --
// block = 256 (8 waves), one point per wave, lane = kernel index (32 kernels)
// writes x0 rows: 0..2 = raw xyz, 3..34 = kc  (rows 35..63 pre-zeroed padding)
__global__ void kc_kernel(const float* __restrict__ pts, const float* __restrict__ kap,
                          const int* __restrict__ knn, float* __restrict__ x0) {
    __shared__ float sk[32 * 16 * 3];
    __shared__ float sk2[32 * 16];
    __shared__ float sdiff[8][16][3];
    __shared__ float sd2[8][16];
    for (int i = threadIdx.x; i < 1536; i += 256) sk[i] = kap[i];
    __syncthreads();
    for (int i = threadIdx.x; i < 512; i += 256) {
        float a = sk[i * 3], b2 = sk[i * 3 + 1], c = sk[i * 3 + 2];
        sk2[i] = a * a + b2 * b2 + c * c;
    }
    const int wave = threadIdx.x >> 5, lane = threadIdx.x & 31;
    const int pt = blockIdx.x * 8 + wave;
    const int b = pt >> 11, n = pt & (NPB - 1);
    const float* P = pts + b * 3 * NPB;
    const float qx = P[n], qy = P[NPB + n], qz = P[2 * NPB + n];
    if (lane < 16) {
        int nb = knn[pt * 17 + 1 + lane];
        float dx = P[nb] - qx, dy = P[NPB + nb] - qy, dz = P[2 * NPB + nb] - qz;
        sdiff[wave][lane][0] = dx; sdiff[wave][lane][1] = dy; sdiff[wave][lane][2] = dz;
        sd2[wave][lane] = dx * dx + dy * dy + dz * dz;
    }
    __syncthreads();
    float acc = 0.f;
    const int kb = lane * 48;   // kernel row base (16 pts * 3)
    for (int j = 0; j < 16; ++j) {
        const float dx = sdiff[wave][j][0], dy = sdiff[wave][j][1], dz = sdiff[wave][j][2];
        const float dj2 = sd2[wave][j];
        #pragma unroll
        for (int m = 0; m < 16; ++m) {
            float kx = sk[kb + m * 3], ky = sk[kb + m * 3 + 1], kz = sk[kb + m * 3 + 2];
            float dd = dj2 + sk2[lane * 16 + m] - 2.f * (dx * kx + dy * ky + dz * kz);
            acc += __expf(dd * -200.0f);   // exp(-dd/0.005)
        }
    }
    x0[(3 + lane) * NCOL + pt] = acc * (1.0f / 16.0f);
    if (lane < 3) x0[lane * NCOL + pt] = P[lane * NPB + n];
}

// -------------------------------------------------------------- WMMA GEMM --
// Y[M,NCOL] = W[M,K] * act(X[Kpad,NCOL]) (+bias), f16 WMMA, f32 accum.
// X must have Kpad rows (rows >= K zero-filled by caller).
// XF (compile-time): act = relu(x*s_in + t_in) (fused BN of previous layer).
// Accumulates per-channel sum / sumsq into stats[2*M].
// POOL (compile-time): atomic max-pool raw outputs into pooled[b*M+m], no Y store.
// Each wave: 16(M) x 64(N) strip = 4 WMMA tiles sharing one A fragment.
// All four B fragments are materialized in distinct registers before the four
// WMMAs issue, so no v_cvt WARs a just-issued WMMA's operands (kills hazard NOPs).
// block 256 (8 waves), grid = (NCOL/512, M/16)
template <bool POOL, bool XF>
__global__ void gemm_bn_kernel(const float* __restrict__ W, const float* __restrict__ bias,
                               const float* __restrict__ X,
                               const float* __restrict__ s_in, const float* __restrict__ t_in,
                               float* __restrict__ Y, float* __restrict__ stats,
                               float* __restrict__ pooled, int M, int K, int Kpad) {
    __shared__ float Wl[192 * 16];           // max Kpad = 192
    __shared__ float sS[192], sT[192];
    const int Mbase = blockIdx.y * 16;
    for (int idx = threadIdx.x; idx < Kpad * 16; idx += 256) {
        int k = idx >> 4, m = idx & 15;
        Wl[idx] = (k < K) ? W[(Mbase + m) * K + k] : 0.f;
    }
    if constexpr (XF) {
        for (int k = threadIdx.x; k < Kpad; k += 256) {
            sS[k] = (k < K) ? s_in[k] : 0.f;
            sT[k] = (k < K) ? t_in[k] : 0.f;
        }
    }
    __syncthreads();

    const int lane = threadIdx.x & 31, wave = threadIdx.x >> 5;
    const int colbase = (blockIdx.x * 8 + wave) * 64;
    const int ncol0 = colbase + (lane & 15);
    const bool hi = lane >= 16;
    const int m = lane & 15;

    v8f acc[4];
    #pragma unroll
    for (int t = 0; t < 4; ++t)
        #pragma unroll
        for (int r = 0; r < 8; ++r) acc[t][r] = 0.f;

    for (int k0 = 0; k0 < Kpad; k0 += 32) {
        // A fragment: 16x32 f16 (weights), shared by 4 tiles
        v16h a;
        const int khA = k0 + (hi ? 16 : 0);
        #pragma unroll
        for (int i = 0; i < 8; ++i) {
            a[i]     = (_Float16)Wl[(khA + i) * 16 + m];
            a[8 + i] = (_Float16)Wl[(khA + 8 + i) * 16 + m];
        }
        const int khB = k0 + (hi ? 16 : 0);
        const float* Xb = X + (size_t)khB * NCOL + ncol0;
        // Build ALL four B fragments first (distinct registers) ...
        v16h bf[4];
        #pragma unroll
        for (int t = 0; t < 4; ++t) {
            #pragma unroll
            for (int i = 0; i < 16; ++i) {
                float v = Xb[(size_t)i * NCOL + t * 16];
                if constexpr (XF) v = fmaxf(fmaf(v, sS[khB + i], sT[khB + i]), 0.f);
                bf[t][i] = (_Float16)v;
            }
        }
        // ... then issue the four independent WMMAs back-to-back
        #pragma unroll
        for (int t = 0; t < 4; ++t)
            acc[t] = __builtin_amdgcn_wmma_f32_16x16x32_f16(false, a, false, bf[t],
                                                            (short)0, acc[t], false, false);
    }

    #pragma unroll
    for (int r = 0; r < 8; ++r) {
        const int gm = Mbase + r + (hi ? 8 : 0);
        const float bia = bias[gm];
        float v0 = acc[0][r] + bia, v1 = acc[1][r] + bia;
        float v2 = acc[2][r] + bia, v3 = acc[3][r] + bia;
        if constexpr (!POOL) {
            float* yr = Y + (size_t)gm * NCOL + ncol0;
            yr[0] = v0; yr[16] = v1; yr[32] = v2; yr[48] = v3;
        }
        float sv = (v0 + v1) + (v2 + v3);
        float sq = (v0 * v0 + v1 * v1) + (v2 * v2 + v3 * v3);
        float mx = fmaxf(fmaxf(v0, v1), fmaxf(v2, v3));
        #pragma unroll
        for (int off = 8; off; off >>= 1) {
            sv += __shfl_xor(sv, off, 32);
            sq += __shfl_xor(sq, off, 32);
            if constexpr (POOL) mx = fmaxf(mx, __shfl_xor(mx, off, 32));
        }
        if ((lane & 15) == 0) {
            atomicAdd(&stats[gm], sv);
            atomicAdd(&stats[M + gm], sq);
            if constexpr (POOL)
                atomicMaxF(&pooled[(colbase >> 11) * M + gm], mx);
        }
    }
}

// ---------------------------------------------------- BN stats -> affine ---
__global__ void finalize_kernel(const float* __restrict__ stats, const float* __restrict__ g,
                                const float* __restrict__ b, float* __restrict__ s,
                                float* __restrict__ t, int M, float invcnt) {
    int m = blockIdx.x * blockDim.x + threadIdx.x;
    if (m >= M) return;
    float mean = stats[m] * invcnt;
    float var  = stats[M + m] * invcnt - mean * mean;
    float sc = g[m] * rsqrtf(var + 1e-5f);
    s[m] = sc;
    t[m] = fmaf(-mean, sc, b[m]);
}

// ------------------------------------------------ elementwise BN+ReLU ------
__global__ void apply_bn_relu_kernel(const float* __restrict__ x, const float* __restrict__ s,
                                     const float* __restrict__ t, float* __restrict__ y, int n) {
    int i = blockIdx.x * blockDim.x + threadIdx.x;
    if (i >= n) return;
    int row = i >> 14;
    y[i] = fmaxf(fmaf(x[i], s[row], t[row]), 0.f);
}

// ------------------------------------------- gather + max over neighbors ---
__global__ void gather_max_kernel(const float* __restrict__ x2, const int* __restrict__ knn,
                                  float* __restrict__ y) {
    const int ch = blockIdx.x;
    const int col = blockIdx.y * 256 + threadIdx.x;
    const int b = col >> 11;
    const float* row = x2 + (size_t)ch * NCOL + (b << 11);
    const int* kn = knn + col * 17;
    float m = NEG_BIG;
    #pragma unroll
    for (int i = 0; i < 17; ++i) m = fmaxf(m, row[kn[i]]);
    y[(size_t)ch * NCOL + col] = m;
}

// ------------------------------------- build conv5 input: [bn4relu(a4); y] --
__global__ void build_x5in_kernel(const float* __restrict__ a4, const float* __restrict__ s4,
                                  const float* __restrict__ t4, const float* __restrict__ y,
                                  float* __restrict__ x5in) {
    int i = blockIdx.x * blockDim.x + threadIdx.x;   // over 192*NCOL
    if (i >= 192 * NCOL) return;
    int row = i >> 14;
    if (row < 128) x5in[i] = fmaxf(fmaf(a4[i], s4[row], t4[row]), 0.f);
    else           x5in[i] = y[i - 128 * NCOL];
}

// -------------------------------------------------------------- FC head ----
// single block, 1024 threads; batch-BN (mean/var over B=8) between FCs
__global__ void head_kernel(const float* __restrict__ pooled, const float* __restrict__ s5,
                            const float* __restrict__ t5,
                            const float* __restrict__ f1w, const float* __restrict__ f1b,
                            const float* __restrict__ g6, const float* __restrict__ b6,
                            const float* __restrict__ f2w, const float* __restrict__ f2b,
                            const float* __restrict__ g7, const float* __restrict__ b7,
                            const float* __restrict__ f3w, const float* __restrict__ f3b,
                            float* __restrict__ out) {
    __shared__ float x5[8 * 1024];
    __shared__ float h1[8 * 512];
    __shared__ float h2[8 * 256];
    const int tid = threadIdx.x;
    for (int i = tid; i < 8192; i += 1024) {
        int ch = i & 1023;
        x5[i] = fmaxf(fmaf(pooled[i], s5[ch], t5[ch]), 0.f);
    }
    __syncthreads();
    for (int i = tid; i < 4096; i += 1024) {
        int b = i >> 9, o = i & 511;
        float s = f1b[o];
        const float* xr = &x5[b * 1024];
        const float* wr = &f1w[o * 1024];
        for (int c = 0; c < 1024; ++c) s = fmaf(xr[c], wr[c], s);
        h1[i] = s;
    }
    __syncthreads();
    if (tid < 512) {
        float mn = 0.f;
        for (int b = 0; b < 8; ++b) mn += h1[b * 512 + tid];
        mn *= 0.125f;
        float var = 0.f;
        for (int b = 0; b < 8; ++b) { float d = h1[b * 512 + tid] - mn; var += d * d; }
        var *= 0.125f;
        float sc = g6[tid] * rsqrtf(var + 1e-5f);
        float sh = fmaf(-mn, sc, b6[tid]);
        for (int b = 0; b < 8; ++b)
            h1[b * 512 + tid] = fmaxf(fmaf(h1[b * 512 + tid], sc, sh), 0.f);
    }
    __syncthreads();
    for (int i = tid; i < 2048; i += 1024) {
        int b = i >> 8, o = i & 255;
        float s = f2b[o];
        const float* xr = &h1[b * 512];
        const float* wr = &f2w[o * 512];
        for (int c = 0; c < 512; ++c) s = fmaf(xr[c], wr[c], s);
        h2[i] = s;
    }
    __syncthreads();
    if (tid < 256) {
        float mn = 0.f;
        for (int b = 0; b < 8; ++b) mn += h2[b * 256 + tid];
        mn *= 0.125f;
        float var = 0.f;
        for (int b = 0; b < 8; ++b) { float d = h2[b * 256 + tid] - mn; var += d * d; }
        var *= 0.125f;
        float sc = g7[tid] * rsqrtf(var + 1e-5f);
        float sh = fmaf(-mn, sc, b7[tid]);
        for (int b = 0; b < 8; ++b)
            h2[b * 256 + tid] = fmaxf(fmaf(h2[b * 256 + tid], sc, sh), 0.f);
    }
    __syncthreads();
    if (tid < 320) {
        int b = tid / 40, o = tid % 40;
        float s = f3b[o];
        const float* xr = &h2[b * 256];
        const float* wr = &f3w[o * 256];
        for (int c = 0; c < 256; ++c) s = fmaf(xr[c], wr[c], s);
        out[b * 40 + o] = s;
    }
}

// ---------------------------------------------------------------------------
extern "C" void kernel_launch(void* const* d_in, const int* in_sizes, int n_in,
                              void* d_out, int out_size, void* d_ws, size_t ws_size,
                              hipStream_t stream) {
    (void)in_sizes; (void)n_in; (void)out_size; (void)ws_size;
    const float* points = (const float*)d_in[0];
    const float* kap    = (const float*)d_in[1];
    const float *c1w=(const float*)d_in[2],  *c1b=(const float*)d_in[3],
                *g1=(const float*)d_in[4],   *b1=(const float*)d_in[5];
    const float *c2w=(const float*)d_in[6],  *c2b=(const float*)d_in[7],
                *g2=(const float*)d_in[8],   *b2=(const float*)d_in[9];
    const float *c3w=(const float*)d_in[10], *c3b=(const float*)d_in[11],
                *g3=(const float*)d_in[12],  *b3=(const float*)d_in[13];
    const float *c4w=(const float*)d_in[14], *c4b=(const float*)d_in[15],
                *g4=(const float*)d_in[16],  *b4=(const float*)d_in[17];
    const float *c5w=(const float*)d_in[18], *c5b=(const float*)d_in[19],
                *g5=(const float*)d_in[20],  *b5=(const float*)d_in[21];
    const float *f1w=(const float*)d_in[22], *f1b=(const float*)d_in[23],
                *g6=(const float*)d_in[24],  *b6=(const float*)d_in[25];
    const float *f2w=(const float*)d_in[26], *f2b=(const float*)d_in[27],
                *g7=(const float*)d_in[28],  *b7=(const float*)d_in[29];
    const float *f3w=(const float*)d_in[30], *f3b=(const float*)d_in[31];
    float* out = (float*)d_out;

    // ---- workspace carve-up -------------------------------------------------
    char* w = (char*)d_ws;
    int* knn = (int*)w;                                  // 8*2048*17 ints
    float* f = (float*)(w + (size_t)8 * 2048 * 17 * 4);
    float* x0    = f; f += 64  * NCOL;                   // 35 used + 29 zero pad
    float* a1    = f; f += 64  * NCOL;
    float* a2    = f; f += 64  * NCOL;
    float* x2act = f; f += 64  * NCOL;
    float* yb    = f; f += 64  * NCOL;
    float* a3    = f; f += 64  * NCOL;
    float* a4    = f; f += 128 * NCOL;
    float* x5in  = f; f += 192 * NCOL;
    float* st1 = f; f += 128;   float* st2 = f; f += 128;
    float* st3 = f; f += 128;   float* st4 = f; f += 256;
    float* st5 = f; f += 2048;                           // stats: contiguous 2688
    float* s1 = f; f += 64;  float* t1 = f; f += 64;
    float* s2 = f; f += 64;  float* t2 = f; f += 64;
    float* s3 = f; f += 64;  float* t3 = f; f += 64;
    float* s4 = f; f += 128; float* t4 = f; f += 128;
    float* s5 = f; f += 1024; float* t5 = f; f += 1024;
    float* pooled = f; f += 8 * 1024;

    const float invc = 1.0f / (float)NCOL;

    // 0) reset accumulators + zero x0 padding rows (35..63); deterministic per call
    fillf_kernel<<<11, 256, 0, stream>>>(st1, 2688, 0.f);
    fillf_kernel<<<32, 256, 0, stream>>>(pooled, 8192, NEG_BIG);
    fillf_kernel<<<1856, 256, 0, stream>>>(x0 + 35 * NCOL, 29 * NCOL, 0.f);

    // 1) KNN (top-17 including self, lowest-index tie-break)
    knn_kernel<<<dim3(512, 8), 128, 0, stream>>>(points, knn);

    // 2) kernel correlation -> x0 rows 0..34
    kc_kernel<<<2048, 256, 0, stream>>>(points, kap, knn, x0);

    // 3) conv1 (no input transform; x0 padded to 64 rows)
    gemm_bn_kernel<false, false><<<dim3(32, 4), 256, 0, stream>>>(
        c1w, c1b, x0, nullptr, nullptr, a1, st1, nullptr, 64, 35, 64);
    finalize_kernel<<<1, 256, 0, stream>>>(st1, g1, b1, s1, t1, 64, invc);

    // 4) conv2 (fused bn1+relu on load)
    gemm_bn_kernel<false, true><<<dim3(32, 4), 256, 0, stream>>>(
        c2w, c2b, a1, s1, t1, a2, st2, nullptr, 64, 64, 64);
    finalize_kernel<<<1, 256, 0, stream>>>(st2, g2, b2, s2, t2, 64, invc);

    // 5) x2act = relu(bn2(a2)) (needed for neighbor gather AND conv3 input)
    apply_bn_relu_kernel<<<4096, 256, 0, stream>>>(a2, s2, t2, x2act, 64 * NCOL);

    // 6) y = max over 17 knn of x2act
    gather_max_kernel<<<dim3(64, 64), 256, 0, stream>>>(x2act, knn, yb);

    // 7) conv3 (input already activated)
    gemm_bn_kernel<false, false><<<dim3(32, 4), 256, 0, stream>>>(
        c3w, c3b, x2act, nullptr, nullptr, a3, st3, nullptr, 64, 64, 64);
    finalize_kernel<<<1, 256, 0, stream>>>(st3, g3, b3, s3, t3, 64, invc);

    // 8) conv4 (fused bn3+relu on load)
    gemm_bn_kernel<false, true><<<dim3(32, 8), 256, 0, stream>>>(
        c4w, c4b, a3, s3, t3, a4, st4, nullptr, 128, 64, 64);
    finalize_kernel<<<1, 256, 0, stream>>>(st4, g4, b4, s4, t4, 128, invc);

    // 9) concat [relu(bn4(a4)); y] -> x5in [192][NCOL]
    build_x5in_kernel<<<12288, 256, 0, stream>>>(a4, s4, t4, yb, x5in);

    // 10) conv5 with fused global max-pool (monotone affine+relu commutes with max)
    gemm_bn_kernel<true, false><<<dim3(32, 64), 256, 0, stream>>>(
        c5w, c5b, x5in, nullptr, nullptr, nullptr, st5, pooled, 1024, 192, 192);
    finalize_kernel<<<4, 256, 0, stream>>>(st5, g5, b5, s5, t5, 1024, invc);

    // 11) FC head (relu(bn5) applied to pooled max, then fc/bn/fc/bn/fc)
    head_kernel<<<1, 1024, 0, stream>>>(pooled, s5, t5, f1w, f1b, g6, b6,
                                        f2w, f2b, g7, b7, f3w, f3b, out);
}
